// Attention_7103875907779
// MI455X (gfx1250) — compile-verified, matrix operand order checked by reading, not run on hardware
//
#include <hip/hip_runtime.h>
#include <hip/hip_bf16.h>
#include <math.h>

// ---------------- problem constants ----------------
#define BB    32
#define TT    2000
#define ENC2  1024
#define DEC   512
#define ATTN  512
#define NK    10
#define KW    100
#define PADC  50

#define MTOT       (BB * TT)      // 64000 flattened rows (b,t)
#define MTILES     (MTOT / 16)    // 4000 16-row tiles (2000 = 125*16: no b-straddle)
#define KTILES     (ENC2 / 32)    // 32 K-tiles of the enc GEMM
#define KTILES_ALL (KTILES + 1)   // +1 fused tile for tmpconv @ W_c2s (K=10 pad 32)
#define NTILES     (ATTN / 16)    // 32

#define MBLK   128                // rows per main-kernel workgroup
#define MAINWG (MTOT / MBLK)      // 500 workgroups

typedef __attribute__((ext_vector_type(16))) __bf16 v16bf;
typedef __attribute__((ext_vector_type(8)))  float  v8f;

union ABu { v16bf v; unsigned u[8]; uint4 q[2]; };

// round-to-nearest-even f32 -> bf16, packed pair into one dword
__device__ __forceinline__ unsigned pk_bf16(float lo, float hi) {
  unsigned a = __float_as_uint(lo);
  unsigned b = __float_as_uint(hi);
  a += 0x7fffu + ((a >> 16) & 1u);
  b += 0x7fffu + ((b >> 16) & 1u);
  return (a >> 16) | (b & 0xffff0000u);
}

// ---------------------------------------------------------------------------
// Kernel 1: swizzle [W_enc ; W_c2s] (f32 row-major [K][N]) into bf16 packed in
// the per-lane WMMA B-register layout:
//   Bsw[((kt*NTILES + nt)*32 + lane)*8 + r]
// lane = n_in_tile + 16h; r<4 -> K pair 8h+2r, r>=4 -> K pair 16+8h+2(r-4).
// ---------------------------------------------------------------------------
__global__ __launch_bounds__(256) void swz_kernel(const float* __restrict__ W_enc,
                                                  const float* __restrict__ W_c2s,
                                                  unsigned* __restrict__ Bsw) {
  int idx = blockIdx.x * 256 + threadIdx.x;
  const int total = KTILES_ALL * NTILES * 32 * 8;
  if (idx >= total) return;
  int r    = idx & 7;
  int lane = (idx >> 3) & 31;
  int nt   = (idx >> 8) & 31;
  int kt   = idx >> 13;
  int n  = nt * 16 + (lane & 15);
  int h  = lane >> 4;
  int kb = (r < 4) ? (8 * h + 2 * r) : (16 + 8 * h + 2 * (r - 4));
  float lo, hi;
  if (kt < KTILES) {
    size_t k = (size_t)kt * 32 + kb;
    lo = W_enc[k * ATTN + n];
    hi = W_enc[(k + 1) * ATTN + n];
  } else {                          // fused conv tile: rows 0..9 = W_c2s
    lo = (kb     < NK) ? W_c2s[kb * ATTN + n]       : 0.f;
    hi = (kb + 1 < NK) ? W_c2s[(kb + 1) * ATTN + n] : 0.f;
  }
  Bsw[idx] = pk_bf16(lo, hi);
}

// ---------------------------------------------------------------------------
// Kernel 2: bias[b][a] = b_enc[a] + sum_d h[b][d] * W_dec[d][a]
// ---------------------------------------------------------------------------
__global__ __launch_bounds__(256) void bias_kernel(const float* __restrict__ hdec,
                                                   const float* __restrict__ W_dec,
                                                   const float* __restrict__ b_enc,
                                                   float* __restrict__ bias) {
  int idx = blockIdx.x * 256 + threadIdx.x;        // BB*ATTN = 16384
  if (idx >= BB * ATTN) return;
  int b = idx >> 9, a = idx & 511;
  float s = b_enc[a];
  const float* hp = hdec + (size_t)b * DEC;
  for (int d = 0; d < DEC; ++d) s += hp[d] * W_dec[(size_t)d * ATTN + a];
  bias[idx] = s;
}

// ---------------------------------------------------------------------------
// Kernel 3: location conv; tconv[(b,t)][k] padded to 16 (k>=10 -> 0)
// ---------------------------------------------------------------------------
__global__ __launch_bounds__(256) void conv_kernel(const float* __restrict__ alpha,
                                                   const float* __restrict__ W_conv,
                                                   float* __restrict__ tconv) {
  int idx = blockIdx.x * 256 + threadIdx.x;        // MTOT*16
  if (idx >= MTOT * 16) return;
  int k   = idx & 15;
  int row = idx >> 4;
  int b = row / TT, t = row - b * TT;
  float s = 0.f;
  if (k < NK) {
    const float* ar = alpha + (size_t)b * TT;
    const float* wr = W_conv + k * KW;
    #pragma unroll 4
    for (int j = 0; j < KW; ++j) {
      int tt = t + j - PADC;
      float x = (tt >= 0 && tt < TT) ? ar[tt] : 0.f;
      s += x * wr[j];
    }
  }
  tconv[idx] = s;
}

// ---------------------------------------------------------------------------
// Kernel 4 (main): fused  score[row] = sum_a w_score[a] *
//        tanh( enc@W_enc + tconv@W_c2s + bias[b,a] )
// Workgroup: 128 rows x 512 cols, 16 waves.
//   wave = (mg, nc): mg = wave>>3 selects 64-row half, nc = wave&7 selects
//   64-col chunk. Per wave: 4 M-tiles x 4 N-tiles -> 128 f32 accum VGPRs.
// A tile (128x32 f32) is converted to bf16 ONCE per workgroup into LDS in the
// WMMA A-register layout, double-buffered; waves fetch it with ds_load_b128.
// ---------------------------------------------------------------------------
__global__ __launch_bounds__(512) void attn_main(const float*    __restrict__ enc,
                                                 const unsigned* __restrict__ Bsw,
                                                 const float*    __restrict__ bias,
                                                 const float*    __restrict__ tconv,
                                                 const float*    __restrict__ wscore,
                                                 float*          __restrict__ score) {
  __shared__ unsigned Abuf[2][8 * 256];   // 8 M-tiles * 256 dwords, bf16 pairs
  __shared__ float    srow[MBLK];

  const int tid  = threadIdx.x;
  const int lane = tid & 31;
  const int wave = tid >> 5;
  const int mg   = wave >> 3;     // 0..1: 64-row half
  const int nc   = wave & 7;      // 0..7: 64-col chunk
  const int m    = lane & 15;
  const int h    = lane >> 4;
  const int mblock = blockIdx.x * MBLK;

  if (tid < MBLK) srow[tid] = 0.f;

  // --- staging mapping: thread -> (row, 8-float segment) -----------------
  const int srow_id = tid >> 2;              // 0..127
  const int seg     = tid & 3;               // segment of 8 floats in K-tile
  const int dmt     = srow_id >> 4;          // dest M-tile
  const int dm      = srow_id & 15;
  const int dlane   = dm + 16 * (seg & 1);   // A-layout lane
  const int dbase   = dmt * 256 + dlane * 8 + ((seg & 2) ? 4 : 0);
  const float* grow = enc  + (size_t)(mblock + srow_id) * ENC2 + seg * 8;
  const float* crow = tconv + (size_t)(mblock + srow_id) * 16  + seg * 8;

  // prologue: stage K-tile 0 into buffer 0
  {
    float4 f0 = *reinterpret_cast<const float4*>(grow);
    float4 f1 = *reinterpret_cast<const float4*>(grow + 4);
    unsigned* d = &Abuf[0][dbase];
    d[0] = pk_bf16(f0.x, f0.y); d[1] = pk_bf16(f0.z, f0.w);
    d[2] = pk_bf16(f1.x, f1.y); d[3] = pk_bf16(f1.z, f1.w);
  }
  __syncthreads();

  v8f acc[4][4] = {};
  const unsigned* bbase = Bsw + (size_t)(nc * 4) * 256 + (size_t)lane * 8;
  const int mtbase = mg * 4;

  for (int kt = 0; kt < KTILES_ALL; ++kt) {
    const int p = kt & 1;

    // A operands for this wave's 4 M-tiles (2x ds_load_b128 each)
    ABu a[4];
    #pragma unroll
    for (int mt = 0; mt < 4; ++mt) {
      const uint4* lp =
          reinterpret_cast<const uint4*>(&Abuf[p][(mtbase + mt) * 256 + lane * 8]);
      a[mt].q[0] = lp[0];
      a[mt].q[1] = lp[1];
    }

    // stage K-tile kt+1 into the other buffer
    if (kt + 1 < KTILES) {
      const float4* ap = reinterpret_cast<const float4*>(grow + (kt + 1) * 32);
      if (kt + 2 < KTILES)
        __builtin_prefetch(grow + (kt + 2) * 32, 0, 1);  // global_prefetch
      float4 f0 = ap[0];
      float4 f1 = ap[1];
      unsigned* d = &Abuf[1 - p][dbase];
      d[0] = pk_bf16(f0.x, f0.y); d[1] = pk_bf16(f0.z, f0.w);
      d[2] = pk_bf16(f1.x, f1.y); d[3] = pk_bf16(f1.z, f1.w);
    } else if (kt + 1 == KTILES) {
      // fused conv tile: tconv rows are 16 floats, zero beyond
      float4 f0 = make_float4(0.f, 0.f, 0.f, 0.f);
      float4 f1 = f0;
      if (seg < 2) {
        f0 = *reinterpret_cast<const float4*>(crow);
        f1 = *reinterpret_cast<const float4*>(crow + 4);
      }
      unsigned* d = &Abuf[1 - p][dbase];
      d[0] = pk_bf16(f0.x, f0.y); d[1] = pk_bf16(f0.z, f0.w);
      d[2] = pk_bf16(f1.x, f1.y); d[3] = pk_bf16(f1.z, f1.w);
    }

    // B tiles + 16 WMMAs (B reloaded per i; A held in regs across i)
    const uint4* bp = reinterpret_cast<const uint4*>(bbase + (size_t)kt * 8192);
    #pragma unroll
    for (int i = 0; i < 4; ++i) {
      ABu bb;
      bb.q[0] = bp[i * 64];
      bb.q[1] = bp[i * 64 + 1];
      #pragma unroll
      for (int mt = 0; mt < 4; ++mt)
        acc[mt][i] = __builtin_amdgcn_wmma_f32_16x16x32_bf16(
            false, a[mt].v, false, bb.v, (short)0, acc[mt][i], false, false);
    }
    __syncthreads();
  }

  // --- epilogue: tanh, weight by w_score, reduce over 512 columns ---------
  // C layout: lane holds col m; VGPR v -> row v + 8h of the tile.
  float part[4][8];
  #pragma unroll
  for (int mt = 0; mt < 4; ++mt)
    #pragma unroll
    for (int v = 0; v < 8; ++v) part[mt][v] = 0.f;

  #pragma unroll
  for (int mt = 0; mt < 4; ++mt) {
    const int mtg  = blockIdx.x * (MBLK / 16) + mtbase + mt;  // global M-tile
    const int bidx = mtg / (TT / 16);                          // batch of tile
    #pragma unroll
    for (int i = 0; i < 4; ++i) {
      const int col = nc * 64 + i * 16 + m;
      const float bs = bias[(size_t)bidx * ATTN + col];
      const float w  = wscore[col];
      #pragma unroll
      for (int v = 0; v < 8; ++v)
        part[mt][v] += w * tanhf(acc[mt][i][v] + bs);
    }
  }
  // butterfly across the 16 lanes sharing the same 8 rows (masks <=8 stay in half)
  #pragma unroll
  for (int off = 8; off >= 1; off >>= 1)
    #pragma unroll
    for (int mt = 0; mt < 4; ++mt)
      #pragma unroll
      for (int v = 0; v < 8; ++v)
        part[mt][v] += __shfl_xor(part[mt][v], off, 32);

  if (m == 0) {
    #pragma unroll
    for (int mt = 0; mt < 4; ++mt)
      #pragma unroll
      for (int v = 0; v < 8; ++v)
        atomicAdd(&srow[(mtbase + mt) * 16 + 8 * h + v], part[mt][v]);
  }
  __syncthreads();
  if (tid < MBLK) score[mblock + tid] = srow[tid];
}

// ---------------------------------------------------------------------------
// Kernel 5: per-batch masked softmax over T
// ---------------------------------------------------------------------------
__global__ __launch_bounds__(256) void softmax_kernel(const float* __restrict__ score,
                                                      const float* __restrict__ mask,
                                                      float* __restrict__ out) {
  __shared__ float red[256];
  int b = blockIdx.x, tid = threadIdx.x;
  const float* sp = score + (size_t)b * TT;
  const float* mp = mask  + (size_t)b * TT;
  float mx = -3.4e38f;
  for (int t = tid; t < TT; t += 256) mx = fmaxf(mx, sp[t]);
  red[tid] = mx;
  __syncthreads();
  for (int s = 128; s > 0; s >>= 1) {
    if (tid < s) red[tid] = fmaxf(red[tid], red[tid + s]);
    __syncthreads();
  }
  float M = red[0];
  __syncthreads();
  float sum = 0.f;
  for (int t = tid; t < TT; t += 256) sum += __expf(sp[t] - M) * mp[t];
  red[tid] = sum;
  __syncthreads();
  for (int s = 128; s > 0; s >>= 1) {
    if (tid < s) red[tid] += red[tid + s];
    __syncthreads();
  }
  float inv = 1.f / red[0];
  for (int t = tid; t < TT; t += 256)
    out[(size_t)b * TT + t] = __expf(sp[t] - M) * mp[t] * inv;
}

// ---------------------------------------------------------------------------
extern "C" void kernel_launch(void* const* d_in, const int* in_sizes, int n_in,
                              void* d_out, int out_size, void* d_ws, size_t ws_size,
                              hipStream_t stream) {
  const float* enc    = (const float*)d_in[0];   // [B,T,ENC2]
  const float* hdec   = (const float*)d_in[1];   // [B,DEC]
  const float* alpha  = (const float*)d_in[2];   // [B,1,T]
  const float* mask   = (const float*)d_in[3];   // [B,T]
  const float* W_conv = (const float*)d_in[4];   // [NK,1,KW]
  const float* W_c2s  = (const float*)d_in[5];   // [NK,ATTN]
  const float* W_enc  = (const float*)d_in[6];   // [ENC2,ATTN]
  const float* b_enc  = (const float*)d_in[7];   // [ATTN]
  const float* W_dec  = (const float*)d_in[8];   // [DEC,ATTN]
  const float* wscore = (const float*)d_in[9];   // [ATTN]
  float* out = (float*)d_out;                    // [B,1,T]

  char* ws = (char*)d_ws;
  size_t off = 0;
  unsigned* Bsw = (unsigned*)(ws + off);
  off += (size_t)KTILES_ALL * NTILES * 32 * 8 * sizeof(unsigned);  // ~1.03 MB
  float* bias = (float*)(ws + off);  off += (size_t)BB * ATTN * sizeof(float);
  float* tconv = (float*)(ws + off); off += (size_t)MTOT * 16 * sizeof(float);
  float* score = (float*)(ws + off); off += (size_t)MTOT * sizeof(float);
  (void)ws_size; (void)in_sizes; (void)n_in; (void)out_size;

  const int swz_total = KTILES_ALL * NTILES * 32 * 8;               // 270336
  swz_kernel <<<(swz_total + 255) / 256, 256, 0, stream>>>(W_enc, W_c2s, Bsw);
  bias_kernel<<<(BB * ATTN + 255) / 256, 256, 0, stream>>>(hdec, W_dec, b_enc, bias);
  conv_kernel<<<(MTOT * 16 + 255) / 256, 256, 0, stream>>>(alpha, W_conv, tconv);
  attn_main  <<<MAINWG, 512, 0, stream>>>(enc, Bsw, bias, tconv, wscore, score);
  softmax_kernel<<<BB, 256, 0, stream>>>(score, mask, out);
}
